// ECN_38130719654485
// MI455X (gfx1250) — matrix operand order repaired and checked by reading, not compiled, not run on hardware
//
#include <hip/hip_runtime.h>

// ---------------- problem constants (match reference) ----------------
#define B_       8
#define N_       2048
#define E_       32768
#define G_       4
#define SE_      64
#define EE_      32
#define H_       64
#define VIN_     160      // 2*SE + EE
#define MLP_     128
#define CENTERS_ 10

// f16 packed-fragment weight offsets inside workspace (in _Float16 elements)
#define OFF_W1A 0
#define OFF_W1B 40960
#define OFF_W2A 57344
#define OFF_W2B 98304
#define OFF_NU1 114688
#define OFF_NU2 122880
#define OFF_P1  126976
#define HW_TOTAL 135168

typedef __attribute__((ext_vector_type(16))) _Float16     v16h;
typedef __attribute__((ext_vector_type(8)))  _Float16     v8h;
typedef __attribute__((ext_vector_type(4)))  _Float16     v4h;
typedef __attribute__((ext_vector_type(8)))  float        v8f;
typedef __attribute__((ext_vector_type(8)))  unsigned int u32x8;
typedef __attribute__((ext_vector_type(4)))  unsigned int u32x4;
typedef __attribute__((ext_vector_type(8)))  int          i32x8;
typedef __attribute__((ext_vector_type(4)))  int          i32x4;

#define WMMA_F16(a, b, c) \
  __builtin_amdgcn_wmma_f32_16x16x32_f16(false, (a), false, (b), (short)0, (c), false, false)

__device__ __forceinline__ float leaky01(float x) { return x > 0.0f ? x : 0.01f * x; }

// A fragment (16x32 f16, M x K) gathered from row-major LDS tile as two
// aligned 16-byte LDS loads (ISA 7.12.2 layout: lanes 0-15 row M=lane,
// K {k0..k0+7, k0+16..k0+23}; lanes 16-31 row M=lane-16, K {+8, +24}).
__device__ __forceinline__ v16h load_a_frag(const _Float16* tile, int ld, int k0, int lane) {
  const int row  = lane & 15;
  const int base = (lane >> 4) ? 8 : 0;
  const _Float16* p = tile + row * ld + k0 + base;
  const v8h lo = *(const v8h*)p;
  const v8h hh = *(const v8h*)(p + 16);
  return __builtin_shufflevector(lo, hh, 0, 1, 2, 3, 4, 5, 6, 7,
                                 8, 9, 10, 11, 12, 13, 14, 15);
}

// B fragment from fragment-order packed weights: one aligned 32-byte load.
__device__ __forceinline__ v16h load_b_frag_pk(const _Float16* wpk, int frag, int lane) {
  return *(const v16h*)(wpk + ((size_t)(frag * 32 + lane) << 4));
}

// B fragment from LDS-resident packed weights (two ds_load_b128).
#define LDS_B_FRAG(arr, frag, lane) (*(const v16h*)&(arr)[((frag) * 32 + (lane)) << 4])

// Zero the rows of A that do not belong to group g.
// Done on dwords: 8 x v_cndmask_b32 instead of 16 x v_cndmask_b16.
__device__ __forceinline__ v16h mask_frag(v16h a, bool on) {
  u32x8 w = __builtin_bit_cast(u32x8, a);
  u32x8 r;
#pragma unroll
  for (int i = 0; i < 8; ++i) r[i] = on ? w[i] : 0u;
  return __builtin_bit_cast(v16h, r);
}

// float4 -> packed 4 x f16 LDS store (one ds_store_b64)
__device__ __forceinline__ void st4h(_Float16* dst, float4 x) {
  v4h h;
  h[0] = (_Float16)x.x; h[1] = (_Float16)x.y;
  h[2] = (_Float16)x.z; h[3] = (_Float16)x.w;
  *(v4h*)dst = h;
}

// Pack one element of a row-major [G*K][ldn] f32 weight into WMMA-B fragment
// order: dst[(((g*Ksteps+ks)*Ntiles+nt)*32 + lane)*16 + i] = W[g][ks*32+(i<8 ? hi*8+i : 16+hi*8+i-8)][nt*16+col]
__device__ __forceinline__ void pack_b_frag(const float* __restrict__ src,
                                            _Float16* __restrict__ dst,
                                            int t, int Ksteps, int Ntiles, int ldn) {
  const int frag = t >> 9;          // / 512
  const int r    = t & 511;
  const int lane = r >> 4;
  const int i    = r & 15;
  const int nt   = frag % Ntiles;
  const int tmp  = frag / Ntiles;
  const int ks   = tmp % Ksteps;
  const int g    = tmp / Ksteps;
  const int col  = lane & 15;
  const int hi   = lane >> 4;
  const int k    = ks * 32 + ((i < 8) ? (hi * 8 + i) : (16 + hi * 8 + (i - 8)));
  dst[t] = (_Float16)src[(size_t)(g * Ksteps * 32 + k) * ldn + nt * 16 + col];
}

// ---------------------------------------------------------------------
// Kernel 1: node embeddings, zero accumulators, fragment-packed f16 weights.
// grid: 4096 x 256 == B*N*SE threads exactly.
// ---------------------------------------------------------------------
__global__ void prep_kernel(const float* __restrict__ sites,
                            const float* __restrict__ se_w, const float* __restrict__ se_b,
                            const float* __restrict__ m1a_w, const float* __restrict__ m1b_w,
                            const float* __restrict__ m2a_w, const float* __restrict__ m2b_w,
                            const float* __restrict__ nu1_w, const float* __restrict__ nu2_w,
                            const float* __restrict__ p1_w,
                            float* __restrict__ s, float* __restrict__ msg,
                            float* __restrict__ pSum, _Float16* __restrict__ hw) {
  const int tid  = blockIdx.x * blockDim.x + threadIdx.x;   // 0 .. B*N*SE-1
  const int node = tid >> 6;
  const int f    = tid & 63;
  s[tid]   = sites[node] * se_w[f] + se_b[f];
  msg[tid] = 0.0f;
  if (tid < B_ * MLP_) pSum[tid] = 0.0f;
  const int t = tid;
  if      (t < 40960)  pack_b_frag(m1a_w, hw + OFF_W1A, t,            5, 4, 64);
  else if (t < 57344)  pack_b_frag(m1b_w, hw + OFF_W1B, t - 40960,    2, 4, 64);
  else if (t < 98304)  pack_b_frag(m2a_w, hw + OFF_W2A, t - 57344,    5, 4, 64);
  else if (t < 114688) pack_b_frag(m2b_w, hw + OFF_W2B, t - 98304,    2, 4, 64);
  else if (t < 122880) pack_b_frag(nu1_w, hw + OFF_NU1, t - 114688,   4, 4, 64);
  else if (t < 126976) pack_b_frag(nu2_w, hw + OFF_NU2, t - 122880,   2, 4, 64);
  else if (t < 135168) pack_b_frag(p1_w,  hw + OFF_P1,  t - 126976,   2, 8, 128);
}

// ---------------------------------------------------------------------
// Kernel 2: fused edge pipeline. One wave32 = one 16-edge tile.
// grouped MLPs via masked-A WMMA accumulation over the 4 groups.
// grid: 4096 blocks x 128 threads (4 waves) == B*E/16 waves exactly.
// ---------------------------------------------------------------------
__global__ __launch_bounds__(128) void edge_msg_kernel(
    const float* __restrict__ s, const float* __restrict__ bonds,
    const int* __restrict__ idx1, const int* __restrict__ idx2,
    const int* __restrict__ uc,
    const float* __restrict__ ee_w, const float* __restrict__ ee_b,
    const _Float16* __restrict__ w1a, const float* __restrict__ b1a,
    const _Float16* __restrict__ w1b, const float* __restrict__ b1b,
    const _Float16* __restrict__ w2a, const float* __restrict__ b2a,
    const _Float16* __restrict__ w2b, const float* __restrict__ b2b,
    const float* __restrict__ att1_w, const float* __restrict__ att1_b,
    const float* __restrict__ att2_w, const float* __restrict__ att2_b,
    float* __restrict__ msg) {
  __shared__ __align__(32) _Float16 vt [4][16][VIN_];
  __shared__ __align__(32) _Float16 ht [4][16][H_];
  __shared__ float    ot1[4][16][H_];
  __shared__ float    ot2[4][16][H_];
  __shared__ int      uct[4][16];
  __shared__ float    attg[4][2][16];

  const int wave   = threadIdx.x >> 5;
  const int lane   = threadIdx.x & 31;
  const int waveId = blockIdx.x * 4 + wave;
  const int b   = waveId >> 11;            // / (E_/16)
  const int e0  = (waveId & 2047) << 4;
  const int row = lane & 15;
  const int hi  = lane >> 4;
  const int col = lane & 15;
  const int e   = e0 + row;

  // hint the packed weight streams into cache (global_prefetch_b8)
  __builtin_prefetch(w1a + (lane << 4), 0, 1);
  __builtin_prefetch(w2a + (lane << 4), 0, 1);

  // ---- stage v = [ s[idx1] | s[idx2] | rbf @ ee_w + ee_b ] as f16 tile ----
  const int i1 = idx1[e];
  const int i2 = idx2[e];
  const float4* s2 = (const float4*)(s + ((size_t)b * N_ + i2) * SE_);
  if (hi == 0) {
    uct[wave][row] = uc[e];
    const float4* s1 = (const float4*)(s + ((size_t)b * N_ + i1) * SE_);
#pragma unroll
    for (int f = 0; f < 16; ++f) st4h(&vt[wave][row][f * 4], s1[f]);
#pragma unroll
    for (int f = 0; f < 4; ++f)  st4h(&vt[wave][row][SE_ + f * 4], s2[f]);
  } else {
#pragma unroll
    for (int f = 4; f < 16; ++f) st4h(&vt[wave][row][SE_ + f * 4], s2[f]);
    const float d = bonds[(size_t)b * E_ + e];
    float r[CENTERS_];
#pragma unroll
    for (int c = 0; c < CENTERS_; ++c) {
      const float mu = (float)c * (10.0f / 9.0f);   // linspace(0,10,10)
      const float t  = d - mu;
      r[c] = __expf(-t * t);                        // WIDTH == 1
    }
#pragma unroll 1
    for (int n4 = 0; n4 < EE_ / 4; ++n4) {
      float4 acc;
      float* av = (float*)&acc;
#pragma unroll
      for (int q = 0; q < 4; ++q) {
        float a = ee_b[n4 * 4 + q];
#pragma unroll
        for (int c = 0; c < CENTERS_; ++c) a += r[c] * ee_w[c * EE_ + n4 * 4 + q];
        av[q] = a;
      }
      st4h(&vt[wave][row][2 * SE_ + n4 * 4], acc);
    }
  }
  __syncthreads();

  const int myg = uct[wave][row];

  // ---- layer 1 of both MLPs: masked [16x160] @ [160x64] per group ----
  v8f c1[4] = {};
  v8f c2[4] = {};
#pragma unroll 1
  for (int g = 0; g < G_; ++g) {
    const bool on = (myg == g);
#pragma unroll 1
    for (int ks = 0; ks < VIN_ / 32; ++ks) {
      const int k0 = ks * 32;
      const v16h a = mask_frag(load_a_frag(&vt[wave][0][0], VIN_, k0, lane), on);
      const int fbase = (g * 5 + ks) * 4;
#pragma unroll
      for (int nt = 0; nt < 4; ++nt) {
        const v16h bf1 = load_b_frag_pk(w1a, fbase + nt, lane);
        c1[nt] = WMMA_F16(a, bf1, c1[nt]);
        const v16h bf2 = load_b_frag_pk(w2a, fbase + nt, lane);
        c2[nt] = WMMA_F16(a, bf2, c2[nt]);
      }
    }
  }

  // h1 = leaky(c1 + b1a[g(row)])  -> LDS (f16) as layer-2 A operand
#pragma unroll
  for (int nt = 0; nt < 4; ++nt)
#pragma unroll
    for (int j = 0; j < 8; ++j) {
      const int m  = j + 8 * hi;               // C layout: M = j + 8*(lane>=16)
      const int gg = uct[wave][m];
      const float v = c1[nt][j] + b1a[gg * H_ + nt * 16 + col];
      ht[wave][m][nt * 16 + col] = (_Float16)leaky01(v);
    }
  __syncthreads();

  // ---- layer 2, MLP #1 ----
  {
    v8f d1[4] = {};
#pragma unroll 1
    for (int g = 0; g < G_; ++g) {
      const bool on = (myg == g);
#pragma unroll 1
      for (int ks = 0; ks < 2; ++ks) {
        const int k0 = ks * 32;
        const v16h a = mask_frag(load_a_frag(&ht[wave][0][0], H_, k0, lane), on);
        const int fbase = (g * 2 + ks) * 4;
#pragma unroll
        for (int nt = 0; nt < 4; ++nt) {
          const v16h bf = load_b_frag_pk(w1b, fbase + nt, lane);
          d1[nt] = WMMA_F16(a, bf, d1[nt]);
        }
      }
    }
#pragma unroll
    for (int nt = 0; nt < 4; ++nt)
#pragma unroll
      for (int j = 0; j < 8; ++j) {
        const int m  = j + 8 * hi;
        const int gg = uct[wave][m];
        ot1[wave][m][nt * 16 + col] = leaky01(d1[nt][j] + b1b[gg * H_ + nt * 16 + col]);
      }
  }
  __syncthreads();

  // h2 = leaky(c2 + b2a[g(row)]) -> reuse ht
#pragma unroll
  for (int nt = 0; nt < 4; ++nt)
#pragma unroll
    for (int j = 0; j < 8; ++j) {
      const int m  = j + 8 * hi;
      const int gg = uct[wave][m];
      const float v = c2[nt][j] + b2a[gg * H_ + nt * 16 + col];
      ht[wave][m][nt * 16 + col] = (_Float16)leaky01(v);
    }
  __syncthreads();

  // ---- layer 2, MLP #2 ----
  {
    v8f d2[4] = {};
#pragma unroll 1
    for (int g = 0; g < G_; ++g) {
      const bool on = (myg == g);
#pragma unroll 1
      for (int ks = 0; ks < 2; ++ks) {
        const int k0 = ks * 32;
        const v16h a = mask_frag(load_a_frag(&ht[wave][0][0], H_, k0, lane), on);
        const int fbase = (g * 2 + ks) * 4;
#pragma unroll
        for (int nt = 0; nt < 4; ++nt) {
          const v16h bf = load_b_frag_pk(w2b, fbase + nt, lane);
          d2[nt] = WMMA_F16(a, bf, d2[nt]);
        }
      }
    }
#pragma unroll
    for (int nt = 0; nt < 4; ++nt)
#pragma unroll
      for (int j = 0; j < 8; ++j) {
        const int m  = j + 8 * hi;
        const int gg = uct[wave][m];
        ot2[wave][m][nt * 16 + col] = leaky01(d2[nt][j] + b2b[gg * H_ + nt * 16 + col]);
      }
  }
  __syncthreads();

  // ---- attention gates: lanes 0-15 -> MLP1, lanes 16-31 -> MLP2 ----
  {
    const float* ot = hi ? &ot2[wave][0][0] : &ot1[wave][0][0];
    const float* aw = hi ? att2_w : att1_w;
    const float  ab = hi ? att2_b[0] : att1_b[0];
    float acc = ab;
    for (int k = 0; k < H_; ++k) acc += ot[row * H_ + k] * aw[k];
    attg[wave][hi][row] = 1.0f / (1.0f + __expf(-acc));
  }
  __syncthreads();

  // ---- gated scatter-add of both messages to receiver nodes ----
  for (int t = lane; t < 16 * H_; t += 32) {
    const int r = t >> 6;
    const int n = t & 63;
    const float val = attg[wave][0][r] * ot1[wave][r][n]
                    + attg[wave][1][r] * ot2[wave][r][n];
    atomicAdd(&msg[((size_t)b * N_ + idx2[e0 + r]) * H_ + n], val);
  }
}

// ---------------------------------------------------------------------
// Kernel 3: node update (+residual) and prediction MLP, WMMA throughout.
// nu1 weights are staged into LDS by the Tensor Data Mover (one D# per
// workgroup, issued by wave 0; tracked by TENSORcnt).
// One wave32 = 16 (b,n) rows. grid: 256 x 128 == B*N/16 waves exactly.
// ---------------------------------------------------------------------
__global__ __launch_bounds__(128) void node_pred_kernel(
    const float* __restrict__ s, const float* __restrict__ msg,
    const _Float16* __restrict__ nu1h, const float* __restrict__ nu1_b,
    const _Float16* __restrict__ nu2h, const float* __restrict__ nu2_b,
    const _Float16* __restrict__ p1h, const float* __restrict__ p1_b,
    float* __restrict__ pSum) {
  __shared__ __align__(32) _Float16 xt[4][16][2 * H_];
  __shared__ __align__(32) _Float16 ht[4][16][H_];
  __shared__ __align__(32) _Float16 st[4][16][H_];
  __shared__ __align__(64) _Float16 wnu1[16 * 32 * 16];   // 16KB packed nu1 fragments

  const int wave   = threadIdx.x >> 5;
  const int lane   = threadIdx.x & 31;
  const int waveId = blockIdx.x * 4 + wave;
  const int r0  = waveId * 16;               // flattened (b,n) base; N_%16==0
  const int b   = r0 >> 11;                  // / N_
  const int row = lane & 15;
  const int hi  = lane >> 4;
  const int col = lane & 15;

  __builtin_prefetch(nu2h + (lane << 4), 0, 1);
  __builtin_prefetch(p1h + (lane << 4), 0, 1);

  // ---- TDM: async copy of the 16KB packed nu1 block into LDS ----
#if __has_builtin(__builtin_amdgcn_tensor_load_to_lds) && __has_builtin(__builtin_amdgcn_s_wait_tensorcnt)
  if (threadIdx.x < 32) {                    // one descriptor per workgroup (wave 0)
    const unsigned long long ga = (unsigned long long)(const void*)nu1h;
    const unsigned ldsa = (unsigned)(size_t)(void*)&wnu1[0];   // flat addr[31:0] == LDS offset
    u32x4 g0;
    g0[0] = 1u;                                            // count=1 user D#, no gather
    g0[1] = ldsa;                                          // lds_addr
    g0[2] = (unsigned)ga;                                  // global_addr[31:0]
    g0[3] = ((unsigned)(ga >> 32) & 0x01FFFFFFu)           // global_addr[56:32]
          | 0x80000000u;                                   // type = 2 ("image")
    i32x8 g1;
    g1[0] = 0x00030000;                                    // data_size = 3 (8 bytes)
    g1[1] = (int)(2048u << 16);                            // tensor_dim0 = 2048 (x8B = 16KB)
    g1[2] = (int)(1u << 16);                               // tensor_dim1 = 1
    g1[3] = (int)(2048u << 16);                            // tile_dim0 = 2048
    g1[4] = 1;                                             // tile_dim1 = 1
    g1[5] = 2048;                                          // tensor_dim0_stride[31:0]
    g1[6] = (int)(2048u << 16);                            // tensor_dim1_stride[15:0]
    g1[7] = 0;
    const i32x4 z4 = {0, 0, 0, 0};
#if defined(__clang_major__) && (__clang_major__ >= 23)
    const i32x8 z8 = {0, 0, 0, 0, 0, 0, 0, 0};
    __builtin_amdgcn_tensor_load_to_lds(g0, g1, z4, z4, z8, 0);
#else
    __builtin_amdgcn_tensor_load_to_lds(g0, g1, z4, z4, 0);
#endif
    __builtin_amdgcn_s_wait_tensorcnt(0);                  // s_wait_tensorcnt 0
  }
#else
  for (int t = (int)threadIdx.x; t < 16 * 32 * 16 / 8; t += (int)blockDim.x)
    ((unsigned long long*)wnu1)[t] = ((const unsigned long long*)nu1h)[t];
#endif

  // stage [ s | msg ] as f16 (vectorized float4 -> 4 x f16)
  const size_t nodeBase = (size_t)(r0 + row) * H_;
  if (hi == 0) {
    const float4* sp = (const float4*)(s + nodeBase);
#pragma unroll
    for (int f = 0; f < 16; ++f) st4h(&xt[wave][row][f * 4], sp[f]);
  } else {
    const float4* mp = (const float4*)(msg + nodeBase);
#pragma unroll
    for (int f = 0; f < 16; ++f) st4h(&xt[wave][row][H_ + f * 4], mp[f]);
  }
  __syncthreads();   // covers both the staging tiles and the TDM-filled wnu1

  // node-update layer 1: [16,128] @ [128,64], B operands from LDS (TDM-staged)
  v8f c[4] = {};
#pragma unroll 1
  for (int ks = 0; ks < 4; ++ks) {
    const int k0 = ks * 32;
    const v16h a = load_a_frag(&xt[wave][0][0], 2 * H_, k0, lane);
#pragma unroll
    for (int nt = 0; nt < 4; ++nt) {
      const v16h bf = LDS_B_FRAG(wnu1, ks * 4 + nt, lane);
      c[nt] = WMMA_F16(a, bf, c[nt]);
    }
  }
#pragma unroll
  for (int nt = 0; nt < 4; ++nt)
#pragma unroll
    for (int j = 0; j < 8; ++j) {
      const int m = j + 8 * hi;
      ht[wave][m][nt * 16 + col] = (_Float16)leaky01(c[nt][j] + nu1_b[nt * 16 + col]);
    }
  __syncthreads();

  // node-update layer 2 + residual: s' = s + leaky(h @ nu2 + b)
  v8f d[4] = {};
#pragma unroll 1
  for (int ks = 0; ks < 2; ++ks) {
    const int k0 = ks * 32;
    const v16h a = load_a_frag(&ht[wave][0][0], H_, k0, lane);
#pragma unroll
    for (int nt = 0; nt < 4; ++nt) {
      const v16h bf = load_b_frag_pk(nu2h, ks * 4 + nt, lane);
      d[nt] = WMMA_F16(a, bf, d[nt]);
    }
  }
#pragma unroll
  for (int nt = 0; nt < 4; ++nt)
#pragma unroll
    for (int j = 0; j < 8; ++j) {
      const int m = j + 8 * hi;
      const float sold = s[(size_t)(r0 + m) * H_ + nt * 16 + col];
      st[wave][m][nt * 16 + col] =
          (_Float16)(sold + leaky01(d[nt][j] + nu2_b[nt * 16 + col]));
    }
  __syncthreads();

  // prediction layer: leaky([16,64] @ [64,128] + b) summed over rows -> pSum[b]
  v8f p[8] = {};
#pragma unroll 1
  for (int ks = 0; ks < 2; ++ks) {
    const int k0 = ks * 32;
    const v16h a = load_a_frag(&st[wave][0][0], H_, k0, lane);
#pragma unroll
    for (int nt = 0; nt < 8; ++nt) {
      const v16h bf = load_b_frag_pk(p1h, ks * 8 + nt, lane);
      p[nt] = WMMA_F16(a, bf, p[nt]);
    }
  }
#pragma unroll
  for (int nt = 0; nt < 8; ++nt) {
    const int mcol = nt * 16 + col;
    const float bias = p1_b[mcol];
    float sum = 0.0f;
#pragma unroll
    for (int j = 0; j < 8; ++j) sum += leaky01(p[nt][j] + bias);
    atomicAdd(&pSum[b * MLP_ + mcol], sum);
  }
}

// ---------------------------------------------------------------------
// Kernel 4: mean over nodes + linear head -> out[B,1]
// ---------------------------------------------------------------------
__global__ void final_kernel(const float* __restrict__ pSum,
                             const float* __restrict__ p2_w,
                             const float* __restrict__ p2_b,
                             float* __restrict__ out) {
  const int b = threadIdx.x;
  if (b >= B_) return;
  float acc = p2_b[0];
  const float invN = 1.0f / (float)N_;
  for (int m = 0; m < MLP_; ++m) acc += (pSum[b * MLP_ + m] * invN) * p2_w[m];
  out[b] = acc;
}

// ---------------------------------------------------------------------
extern "C" void kernel_launch(void* const* d_in, const int* in_sizes, int n_in,
                              void* d_out, int out_size, void* d_ws, size_t ws_size,
                              hipStream_t stream) {
  const float* sites  = (const float*)d_in[0];
  const float* bonds  = (const float*)d_in[1];
  const int*   idx1   = (const int*)d_in[2];
  const int*   idx2   = (const int*)d_in[3];
  const int*   uc     = (const int*)d_in[4];
  const float* se_w   = (const float*)d_in[5];
  const float* se_b   = (const float*)d_in[6];
  const float* ee_w   = (const float*)d_in[7];
  const float* ee_b   = (const float*)d_in[8];
  const float* m1a_w  = (const float*)d_in[9];
  const float* m1a_b  = (const float*)d_in[10];
  const float* m1b_w  = (const float*)d_in[11];
  const float* m1b_b  = (const float*)d_in[12];
  const float* m2a_w  = (const float*)d_in[13];
  const float* m2a_b  = (const float*)d_in[14];
  const float* m2b_w  = (const float*)d_in[15];
  const float* m2b_b  = (const float*)d_in[16];
  const float* att1_w = (const float*)d_in[17];
  const float* att1_b = (const float*)d_in[18];
  const float* att2_w = (const float*)d_in[19];
  const float* att2_b = (const float*)d_in[20];
  const float* nu1_w  = (const float*)d_in[21];
  const float* nu1_b  = (const float*)d_in[22];
  const float* nu2_w  = (const float*)d_in[23];
  const float* nu2_b  = (const float*)d_in[24];
  const float* p1_w   = (const float*)d_in[25];
  const float* p1_b   = (const float*)d_in[26];
  const float* p2_w   = (const float*)d_in[27];
  const float* p2_b   = (const float*)d_in[28];

  // workspace layout: s (4MB) | msg (4MB) | pSum (4KB) | packed f16 weights (~264KB)
  char* ws = (char*)d_ws;
  float*    sbuf = (float*)ws;
  float*    msg  = (float*)(ws + (size_t)4 * 1048576);
  float*    pSum = (float*)(ws + (size_t)8 * 1048576);
  _Float16* hw   = (_Float16*)(ws + (size_t)8 * 1048576 + 4096);

  const _Float16* w1a  = hw + OFF_W1A;
  const _Float16* w1b  = hw + OFF_W1B;
  const _Float16* w2a  = hw + OFF_W2A;
  const _Float16* w2b  = hw + OFF_W2B;
  const _Float16* nu1h = hw + OFF_NU1;
  const _Float16* nu2h = hw + OFF_NU2;
  const _Float16* p1h  = hw + OFF_P1;

  // 1) embeddings + zeroing + fragment-packed f16 weight staging
  prep_kernel<<<4096, 256, 0, stream>>>(sites, se_w, se_b,
                                        m1a_w, m1b_w, m2a_w, m2b_w,
                                        nu1_w, nu2_w, p1_w,
                                        sbuf, msg, pSum, hw);

  // 2) fused edge pipeline: B*E/16 = 16384 wave32 tiles -> 4096 blocks x 4 waves
  edge_msg_kernel<<<4096, 128, 0, stream>>>(sbuf, bonds, idx1, idx2, uc,
                                            ee_w, ee_b,
                                            w1a, m1a_b, w1b, m1b_b,
                                            w2a, m2a_b, w2b, m2b_b,
                                            att1_w, att1_b, att2_w, att2_b,
                                            msg);

  // 3) node update + prediction: B*N/16 = 1024 waves -> 256 blocks x 4 waves
  node_pred_kernel<<<256, 128, 0, stream>>>(sbuf, msg,
                                            nu1h, nu1_b, nu2h, nu2_b,
                                            p1h, p1_b, pSum);

  // 4) mean pool + head
  final_kernel<<<1, 32, 0, stream>>>(pSum, p2_w, p2_b, (float*)d_out);
}
